// RepPointsHead_11982958756192
// MI455X (gfx1250) — matrix operand order repaired and checked by reading, not compiled, not run on hardware
//
#include <hip/hip_runtime.h>
#include <hip/hip_bf16.h>

// ---------------------------------------------------------------------------
// RepPointsHead forward on gfx1250 (MI455X).  bf16 WMMA implicit-GEMM convs,
// f32 accumulate + f32 GroupNorm stats.  All kernels launched on `stream`.
// Conv GEMM: block tile 128x128, 8 waves x (32Mx64N), double-buffered LDS.
// A/B panels stream via GLOBAL_LOAD_ASYNC_TO_LDS (ASYNCcnt) when the builtin
// is available; otherwise register-staged global loads (round-2 fallback).
// ---------------------------------------------------------------------------

typedef __attribute__((ext_vector_type(16))) __bf16 v16bf;
typedef __attribute__((ext_vector_type(8)))  __bf16 v8bf;
typedef __attribute__((ext_vector_type(8)))  float  v8f;
typedef __attribute__((ext_vector_type(8)))  unsigned short us8;

#define B_    4
#define C_    256
#define H_    64
#define W_    64
#define NSP   (B_ * H_ * W_)      // 16384 spatial positions (all batches)
#define KTOT  (9 * C_)            // 2304 = im2col K for 3x3 conv
#define NSTEP (KTOT / 32)         // 72 K-steps
#define OUTC  116                 // 80 cls + 18 init + 18 refine
#define WSLOT (KTOT * C_)         // ushorts per packed conv weight

#if defined(__has_builtin)
#if __has_builtin(__builtin_amdgcn_global_load_async_to_lds_b128)
#define ASYNC_LDS 1
#endif
#endif
#ifndef ASYNC_LDS
#define ASYNC_LDS 0
#endif

#if ASYNC_LDS
#if __has_builtin(__builtin_amdgcn_s_wait_asynccnt)
#define WAIT_ASYNC0() __builtin_amdgcn_s_wait_asynccnt(0)
#else
#define WAIT_ASYNC0() asm volatile("s_wait_asynccnt 0x0" ::: "memory")
#endif
typedef int v4i __attribute__((vector_size(16)));
typedef __attribute__((address_space(1))) v4i gv4i;   // __device__ int4
typedef __attribute__((address_space(3))) v4i lv4i;   // __shared__ int4
__device__ __forceinline__ void async_b128(const void* g, void* l) {
  gv4i* gp = reinterpret_cast<gv4i*>(reinterpret_cast<uintptr_t>(g));
  lv4i* lp = reinterpret_cast<lv4i*>((unsigned int)reinterpret_cast<uintptr_t>(l));
  __builtin_amdgcn_global_load_async_to_lds_b128(gp, lp, 0, 0);
}
#endif

__device__ __forceinline__ unsigned short f2bf(float f) {
  unsigned int u = __float_as_uint(f);
  u += 0x7FFFu + ((u >> 16) & 1u);          // round-to-nearest-even
  return (unsigned short)(u >> 16);
}
__device__ __forceinline__ float bf2f(unsigned short s) {
  return __uint_as_float(((unsigned int)s) << 16);
}

union FragAB { v16bf v; v8bf h[2]; };
union AccU   { v8f v; float f[8]; };

// ---------------------------------------------------------------------------
// Layout/packing kernels
// ---------------------------------------------------------------------------
__global__ void zero_pad(unsigned short* __restrict__ z) { z[threadIdx.x] = 0; }

__global__ __launch_bounds__(256) void nchw2nhwc_bf16(const float* __restrict__ in,
                                                      unsigned short* __restrict__ out) {
  int i  = blockIdx.x * 256 + threadIdx.x;        // over B*H*W*C
  int n  = i >> 8;                                // global spatial index
  int c  = i & 255;
  int b  = n >> 12;
  int sp = n & 4095;
  out[i] = f2bf(in[(b * C_ + c) * 4096 + sp]);
}

// w: [256,256,3,3] f32  ->  wp: [o][tap*256 + c] bf16
__global__ __launch_bounds__(256) void pack_w(const float* __restrict__ w,
                                              unsigned short* __restrict__ wp) {
  int t   = blockIdx.x * 256 + threadIdx.x;       // over 256*2304
  int o   = t / KTOT;
  int r   = t - o * KTOT;
  int tap = r >> 8;
  int c   = r & 255;
  wp[t] = f2bf(w[(o * C_ + c) * 9 + tap]);
}

// ---------------------------------------------------------------------------
// Implicit-GEMM 3x3 conv, SAME padding.  M=256 (outC), K=2304, N=16384.
// mode 0: raw f32 NHWC to out32 (GroupNorm follows)
// mode 1: +bias, ReLU, bf16 NHWC to outbf
// ---------------------------------------------------------------------------
__global__ __launch_bounds__(256) void conv_gemm(const unsigned short* __restrict__ in,
                                                 const unsigned short* __restrict__ wp,
                                                 float* __restrict__ out32,
                                                 unsigned short* __restrict__ outbf,
                                                 const float* __restrict__ bias,
                                                 const unsigned short* __restrict__ zpad,
                                                 int mode) {
  __shared__ __align__(64) unsigned short sA[2][128][32];
  __shared__ __align__(64) unsigned short sB[2][128][32];

  const int tid  = threadIdx.x;
  const int lane = tid & 31;
  const int wv   = tid >> 5;
  const int wm   = wv >> 1;      // 0..3 (M)
  const int wn   = wv & 1;       // 0..1 (N)
  const int m0   = blockIdx.y * 128;
  const int n0   = blockIdx.x * 128;

  AccU acc[2][4];
#pragma unroll
  for (int a = 0; a < 2; ++a)
#pragma unroll
    for (int b = 0; b < 4; ++b)
#pragma unroll
      for (int v = 0; v < 8; ++v) acc[a][b].f[v] = 0.0f;

  // loader mapping: thread -> (row 0..127, 16-wide K half)
  const int ld   = tid >> 1;               // A weight row / B column
  const int half = tid & 1;
  const int n    = n0 + ld;
  const int bb   = n >> 12;
  const int yy   = (n >> 6) & 63;
  const int xx   = n & 63;
  const unsigned int aRowBase = (unsigned)(m0 + ld) * KTOT + (half << 4);

  const int klo  = (lane >> 4) * 8;        // unused; A-frag also reads at koff
  const int koff = (lane >> 4) * 16;       // B-frag K sub-offset
  const int lm   = lane & 15;

  auto bPtr = [&](int st) -> const unsigned short* {
    const int tap = st >> 3;
    const int dyy = (tap * 11) >> 5;       // tap / 3 for tap in 0..8
    const int dxx = tap - dyy * 3;
    const int sy  = yy + dyy - 1;
    const int sx  = xx + dxx - 1;
    const int c0  = ((st & 7) << 5) + (half << 4);
    if ((unsigned)sy < 64u && (unsigned)sx < 64u)
      return &in[(((bb * 64 + sy) * 64) + sx) * C_ + c0];
    return zpad;                           // SAME padding: zero page
  };

#if ASYNC_LDS
  auto issueAB = [&](int st, int buf) {
    const unsigned short* pA = &wp[aRowBase + (st << 5)];
    async_b128(pA,     &sA[buf][ld][half * 16]);
    async_b128(pA + 8, &sA[buf][ld][half * 16 + 8]);
    const unsigned short* pB = bPtr(st);
    async_b128(pB,     &sB[buf][ld][half * 16]);
    async_b128(pB + 8, &sB[buf][ld][half * 16 + 8]);
  };
  issueAB(0, 0);
#else
  us8 ra0, ra1, rb0, rb1;
  auto loadAB = [&](int st) {
    const unsigned short* pA = &wp[aRowBase + (st << 5)];
    ra0 = *(const us8*)pA;
    ra1 = *(const us8*)(pA + 8);
    const unsigned short* pB = bPtr(st);
    rb0 = *(const us8*)pB;
    rb1 = *(const us8*)(pB + 8);
  };
  auto store = [&](int buf) {
    *(us8*)&sA[buf][ld][half * 16]     = ra0;
    *(us8*)&sA[buf][ld][half * 16 + 8] = ra1;
    *(us8*)&sB[buf][ld][half * 16]     = rb0;
    *(us8*)&sB[buf][ld][half * 16 + 8] = rb1;
  };
  loadAB(0); store(0);
#endif

  for (int st = 0; st < NSTEP; ++st) {
    const int buf   = st & 1;
    const bool more = (st + 1) < NSTEP;
#if ASYNC_LDS
    WAIT_ASYNC0();                         // my writes into LDS[buf] landed
    __syncthreads();                       // everyone's landed; buf^1 free
    if (more) issueAB(st + 1, buf ^ 1);    // DMA flies under the WMMAs
#else
    if (more) loadAB(st + 1);              // in flight during WMMAs
    __syncthreads();                       // LDS[buf] ready
#endif

    FragAB af[2];
#pragma unroll
    for (int tm = 0; tm < 2; ++tm) {
      const int mr = wm * 32 + tm * 16 + lm;
      af[tm].h[0] = *(const v8bf*)&sA[buf][mr][koff];
      af[tm].h[1] = *(const v8bf*)&sA[buf][mr][koff + 8];
    }
#pragma unroll
    for (int tn = 0; tn < 4; ++tn) {
      const int nr = wn * 64 + tn * 16 + lm;
      FragAB bfg;
      bfg.v = *(const v16bf*)&sB[buf][nr][koff];
#pragma unroll
      for (int tm = 0; tm < 2; ++tm) {
        acc[tm][tn].v = __builtin_amdgcn_wmma_f32_16x16x32_bf16(
            false, af[tm].v, false, bfg.v, (short)0, acc[tm][tn].v, false, false);
      }
    }
#if !ASYNC_LDS
    if (more) store(buf ^ 1);
#endif
  }

  // ---- epilogue ----
#pragma unroll
  for (int tm = 0; tm < 2; ++tm) {
#pragma unroll
    for (int tn = 0; tn < 4; ++tn) {
      const int ocb = m0 + wm * 32 + tm * 16 + (lane >> 4) * 8;
      const int nn  = n0 + wn * 64 + tn * 16 + lm;
      if (mode == 0) {
        float* dst = &out32[nn * C_ + ocb];
        *(float4*)(dst)     = *(const float4*)&acc[tm][tn].f[0];
        *(float4*)(dst + 4) = *(const float4*)&acc[tm][tn].f[4];
      } else {
#pragma unroll
        for (int v = 0; v < 8; ++v) {
          float val = acc[tm][tn].f[v] + (bias ? bias[ocb + v] : 0.0f);
          outbf[nn * C_ + ocb + v] = f2bf(val > 0.0f ? val : 0.0f);
        }
      }
    }
  }
}

// ---------------------------------------------------------------------------
// Deformable conv as GEMM: staged bilinear gather feeds the B panel (VALU
// blend), weight panel streams async to LDS.  Out-of-range corners contribute
// zero (weight zeroed, index clamped).  Epilogue: ReLU -> bf16 NHWC.
// ---------------------------------------------------------------------------
__global__ __launch_bounds__(256) void deform_gemm(const unsigned short* __restrict__ in,
                                                   const unsigned short* __restrict__ wp,
                                                   const float* __restrict__ pts,
                                                   unsigned short* __restrict__ outbf) {
  __shared__ __align__(64) unsigned short sA[2][128][32];
  __shared__ __align__(64) unsigned short sB[2][128][32];

  const int tid  = threadIdx.x;
  const int lane = tid & 31;
  const int wv   = tid >> 5;
  const int wm   = wv >> 1;
  const int wn   = wv & 1;
  const int m0   = blockIdx.y * 128;
  const int n0   = blockIdx.x * 128;

  AccU acc[2][4];
#pragma unroll
  for (int a = 0; a < 2; ++a)
#pragma unroll
    for (int b = 0; b < 4; ++b)
#pragma unroll
      for (int v = 0; v < 8; ++v) acc[a][b].f[v] = 0.0f;

  const int ld   = tid >> 1;
  const int half = tid & 1;
  const int n    = n0 + ld;
  const int bb   = n >> 12;
  const int yy   = (n >> 6) & 63;
  const int xx   = n & 63;
  const unsigned int aRowBase = (unsigned)(m0 + ld) * KTOT + (half << 4);

  const int klo  = (lane >> 4) * 8;
  const int koff = (lane >> 4) * 16;
  const int lm   = lane & 15;

  us8 cr[8];          // 4 corners x 16 channels
  float w4[4];

#if ASYNC_LDS
  auto issueA = [&](int st, int buf) {
    const unsigned short* pA = &wp[aRowBase + (st << 5)];
    async_b128(pA,     &sA[buf][ld][half * 16]);
    async_b128(pA + 8, &sA[buf][ld][half * 16 + 8]);
  };
#else
  us8 ra0, ra1;
  auto loadA = [&](int st) {
    const unsigned short* pA = &wp[aRowBase + (st << 5)];
    ra0 = *(const us8*)pA;
    ra1 = *(const us8*)(pA + 8);
  };
  auto storeA = [&](int buf) {
    *(us8*)&sA[buf][ld][half * 16]     = ra0;
    *(us8*)&sA[buf][ld][half * 16 + 8] = ra1;
  };
#endif
  auto loadB = [&](int st) {
    const int tap = st >> 3;
    const int c0  = ((st & 7) << 5) + (half << 4);
    const float py = (float)yy + pts[n * 18 + 2 * tap];
    const float px = (float)xx + pts[n * 18 + 2 * tap + 1];
    const float y0 = floorf(py), x0 = floorf(px);
    const float wy = py - y0, wx = px - x0;
    const int iy0 = (int)y0, ix0 = (int)x0;
    const int cy[4] = {iy0, iy0, iy0 + 1, iy0 + 1};
    const int cx[4] = {ix0, ix0 + 1, ix0, ix0 + 1};
    const float cw[4] = {(1.f - wy) * (1.f - wx), (1.f - wy) * wx,
                         wy * (1.f - wx),         wy * wx};
#pragma unroll
    for (int q = 0; q < 4; ++q) {
      const bool ok = (unsigned)cy[q] < 64u && (unsigned)cx[q] < 64u;
      w4[q] = ok ? cw[q] : 0.0f;
      const int ycl = min(max(cy[q], 0), 63);
      const int xcl = min(max(cx[q], 0), 63);
      const unsigned short* pB = &in[(((bb * 64 + ycl) * 64) + xcl) * C_ + c0];
      cr[2 * q]     = *(const us8*)pB;
      cr[2 * q + 1] = *(const us8*)(pB + 8);
    }
  };
  auto storeB = [&](int buf) {
    us8 b0, b1;
#pragma unroll
    for (int j = 0; j < 8; ++j) {
      float v0 = w4[0] * bf2f(cr[0][j]) + w4[1] * bf2f(cr[2][j]) +
                 w4[2] * bf2f(cr[4][j]) + w4[3] * bf2f(cr[6][j]);
      float v1 = w4[0] * bf2f(cr[1][j]) + w4[1] * bf2f(cr[3][j]) +
                 w4[2] * bf2f(cr[5][j]) + w4[3] * bf2f(cr[7][j]);
      b0[j] = f2bf(v0);
      b1[j] = f2bf(v1);
    }
    *(us8*)&sB[buf][ld][half * 16]     = b0;
    *(us8*)&sB[buf][ld][half * 16 + 8] = b1;
  };

#if ASYNC_LDS
  loadB(0); issueA(0, 0); storeB(0);
#else
  loadB(0); loadA(0); storeA(0); storeB(0);
#endif

  for (int st = 0; st < NSTEP; ++st) {
    const int buf   = st & 1;
    const bool more = (st + 1) < NSTEP;
    if (more) loadB(st + 1);               // corner gathers in flight
#if ASYNC_LDS
    WAIT_ASYNC0();
    __syncthreads();
    if (more) issueA(st + 1, buf ^ 1);
#else
    if (more) loadA(st + 1);
    __syncthreads();
#endif

    FragAB af[2];
#pragma unroll
    for (int tm = 0; tm < 2; ++tm) {
      const int mr = wm * 32 + tm * 16 + lm;
      af[tm].h[0] = *(const v8bf*)&sA[buf][mr][koff];
      af[tm].h[1] = *(const v8bf*)&sA[buf][mr][koff + 8];
    }
#pragma unroll
    for (int tn = 0; tn < 4; ++tn) {
      const int nr = wn * 64 + tn * 16 + lm;
      FragAB bfg;
      bfg.v = *(const v16bf*)&sB[buf][nr][koff];
#pragma unroll
      for (int tm = 0; tm < 2; ++tm) {
        acc[tm][tn].v = __builtin_amdgcn_wmma_f32_16x16x32_bf16(
            false, af[tm].v, false, bfg.v, (short)0, acc[tm][tn].v, false, false);
      }
    }
#if ASYNC_LDS
    if (more) storeB(buf ^ 1);
#else
    if (more) { storeA(buf ^ 1); storeB(buf ^ 1); }
#endif
  }

#pragma unroll
  for (int tm = 0; tm < 2; ++tm) {
#pragma unroll
    for (int tn = 0; tn < 4; ++tn) {
      const int ocb = m0 + wm * 32 + tm * 16 + (lane >> 4) * 8;
      const int nn  = n0 + wn * 64 + tn * 16 + lm;
#pragma unroll
      for (int v = 0; v < 8; ++v) {
        float val = acc[tm][tn].f[v];
        outbf[nn * C_ + ocb + v] = f2bf(val > 0.0f ? val : 0.0f);
      }
    }
  }
}

// ---------------------------------------------------------------------------
// GroupNorm: stats (mean, rstd) per (batch, group) then apply + ReLU -> bf16
// ---------------------------------------------------------------------------
__global__ __launch_bounds__(256) void gn_stats(const float* __restrict__ x,
                                                float* __restrict__ stats) {
  const int bg = blockIdx.x;            // b*32 + g
  const int b  = bg >> 5;
  const int g  = bg & 31;
  __shared__ float ss[256], sq[256];
  float s = 0.f, q = 0.f;
  for (int i = threadIdx.x; i < 4096 * 8; i += 256) {
    const int sp = i >> 3;
    const int c  = (g << 3) + (i & 7);
    const float v = x[(b * 4096 + sp) * C_ + c];
    s += v; q += v * v;
  }
  ss[threadIdx.x] = s; sq[threadIdx.x] = q;
  __syncthreads();
  for (int o = 128; o > 0; o >>= 1) {
    if (threadIdx.x < o) {
      ss[threadIdx.x] += ss[threadIdx.x + o];
      sq[threadIdx.x] += sq[threadIdx.x + o];
    }
    __syncthreads();
  }
  if (threadIdx.x == 0) {
    const float inv  = 1.0f / 32768.0f;
    const float mean = ss[0] * inv;
    const float var  = sq[0] * inv - mean * mean;
    stats[bg * 2]     = mean;
    stats[bg * 2 + 1] = rsqrtf(var + 1e-5f);
  }
}

__global__ __launch_bounds__(256) void gn_apply(const float* __restrict__ x,
                                                const float* __restrict__ gw,
                                                const float* __restrict__ gb,
                                                const float* __restrict__ stats,
                                                unsigned short* __restrict__ out) {
  const int i = blockIdx.x * 256 + threadIdx.x;   // over NSP*C
  const int nn = i >> 8;
  const int c  = i & 255;
  const int b  = nn >> 12;
  const int g  = c >> 3;
  const float mean = stats[(b * 32 + g) * 2];
  const float rstd = stats[(b * 32 + g) * 2 + 1];
  float v = (x[i] - mean) * rstd * gw[c] + gb[c];
  out[i] = f2bf(v > 0.0f ? v : 0.0f);
}

// ---------------------------------------------------------------------------
// 1x1 conv head: bf16 NHWC in, f32 weights [OC,256], writes NCHW d_out slab.
// Optional: store result to pts buffer (init head) or add pts (refine head).
// ---------------------------------------------------------------------------
__global__ __launch_bounds__(256) void conv1x1(const unsigned short* __restrict__ in,
                                               const float* __restrict__ w,
                                               const float* __restrict__ bias,
                                               float* __restrict__ out,
                                               int chanoff, int OC,
                                               const float* __restrict__ pts_add,
                                               float* __restrict__ pts_store) {
  const int nn = blockIdx.x * 256 + threadIdx.x;   // 0..NSP-1
  const int oc = blockIdx.y;
  const unsigned short* ip = &in[nn * C_];
  const float* wr = &w[oc * C_];
  float s = bias[oc];
  for (int c = 0; c < C_; c += 8) {
    const us8 d = *(const us8*)&ip[c];
#pragma unroll
    for (int j = 0; j < 8; ++j) s += bf2f(d[j]) * wr[c + j];
  }
  if (pts_store) pts_store[nn * 18 + oc] = s;
  if (pts_add)   s += pts_add[nn * 18 + oc];
  const int b  = nn >> 12;
  const int sp = nn & 4095;
  out[(b * OUTC + chanoff + oc) * 4096 + sp] = s;
}

// ---------------------------------------------------------------------------
// Host launch
// ---------------------------------------------------------------------------
extern "C" void kernel_launch(void* const* d_in, const int* in_sizes, int n_in,
                              void* d_out, int out_size, void* d_ws, size_t ws_size,
                              hipStream_t stream) {
  const float* features    = (const float*)d_in[0];
  const float* cls_w       = (const float*)d_in[1];
  const float* reg_w       = (const float*)d_in[2];
  const float* cls_gn_w    = (const float*)d_in[3];
  const float* cls_gn_b    = (const float*)d_in[4];
  const float* reg_gn_w    = (const float*)d_in[5];
  const float* reg_gn_b    = (const float*)d_in[6];
  const float* init_conv_w = (const float*)d_in[7];
  const float* init_conv_b = (const float*)d_in[8];
  const float* init_out_w  = (const float*)d_in[9];
  const float* init_out_b  = (const float*)d_in[10];
  const float* cls_dcn_w   = (const float*)d_in[11];
  const float* cls_out_w   = (const float*)d_in[12];
  const float* cls_out_b   = (const float*)d_in[13];
  const float* ref_dcn_w   = (const float*)d_in[14];
  const float* ref_out_w   = (const float*)d_in[15];
  const float* ref_out_b   = (const float*)d_in[16];
  float* out = (float*)d_out;

  // workspace layout
  const size_t ACT = (size_t)NSP * C_ * sizeof(unsigned short);   // 8.39 MB
  char* p = (char*)d_ws;
  unsigned short* featN = (unsigned short*)p; p += ACT;
  unsigned short* bufA  = (unsigned short*)p; p += ACT;
  unsigned short* bufB  = (unsigned short*)p; p += ACT;
  unsigned short* bufC  = (unsigned short*)p; p += ACT;
  unsigned short* bufD  = (unsigned short*)p; p += ACT;
  float* conv32 = (float*)p;          p += (size_t)NSP * C_ * sizeof(float);
  unsigned short* wpack = (unsigned short*)p; p += (size_t)9 * WSLOT * sizeof(unsigned short);
  float* pts   = (float*)p;           p += (size_t)NSP * 18 * sizeof(float);
  float* stats = (float*)p;           p += 1024;
  unsigned short* zpad = (unsigned short*)p;   // 64 ushorts, zeroed below

  const dim3 gemm_grid(NSP / 128, 2);   // 128 N-tiles x 2 M-tiles
  const int  wElems = 256 * KTOT;       // 589824 per conv

  // 0. zero page for SAME-padding async loads
  zero_pad<<<1, 64, 0, stream>>>(zpad);

  // 1. features NCHW f32 -> NHWC bf16
  nchw2nhwc_bf16<<<(NSP * C_) / 256, 256, 0, stream>>>(features, featN);

  // 2. repack all 9 conv weights to bf16 [o][tap*256+c]
  for (int i = 0; i < 3; ++i) {
    pack_w<<<wElems / 256, 256, 0, stream>>>(cls_w + (size_t)i * wElems, wpack + (size_t)i * WSLOT);
    pack_w<<<wElems / 256, 256, 0, stream>>>(reg_w + (size_t)i * wElems, wpack + (size_t)(3 + i) * WSLOT);
  }
  pack_w<<<wElems / 256, 256, 0, stream>>>(init_conv_w, wpack + (size_t)6 * WSLOT);
  pack_w<<<wElems / 256, 256, 0, stream>>>(cls_dcn_w,  wpack + (size_t)7 * WSLOT);
  pack_w<<<wElems / 256, 256, 0, stream>>>(ref_dcn_w,  wpack + (size_t)8 * WSLOT);

  // 3. conv + GN + ReLU chains
  auto convgn = [&](const unsigned short* inp, int slot, const float* gw,
                    const float* gb, unsigned short* outp) {
    conv_gemm<<<gemm_grid, 256, 0, stream>>>(inp, wpack + (size_t)slot * WSLOT,
                                             conv32, nullptr, nullptr, zpad, 0);
    gn_stats<<<128, 256, 0, stream>>>(conv32, stats);
    gn_apply<<<(NSP * C_) / 256, 256, 0, stream>>>(conv32, gw, gb, stats, outp);
  };
  convgn(featN, 0, cls_gn_w + 0,   cls_gn_b + 0,   bufA);
  convgn(bufA,  1, cls_gn_w + 256, cls_gn_b + 256, bufB);
  convgn(bufB,  2, cls_gn_w + 512, cls_gn_b + 512, bufA);   // cls_feat -> bufA
  convgn(featN, 3, reg_gn_w + 0,   reg_gn_b + 0,   bufC);
  convgn(bufC,  4, reg_gn_w + 256, reg_gn_b + 256, bufD);
  convgn(bufD,  5, reg_gn_w + 512, reg_gn_b + 512, bufC);   // reg_feat -> bufC

  // 4. init head: relu(conv3x3(reg_feat)+b) -> bufD ; pts_init -> pts + d_out[80:98]
  conv_gemm<<<gemm_grid, 256, 0, stream>>>(bufC, wpack + (size_t)6 * WSLOT,
                                           nullptr, bufD, init_conv_b, zpad, 1);
  conv1x1<<<dim3(NSP / 256, 18), 256, 0, stream>>>(bufD, init_out_w, init_out_b,
                                                   out, 80, 18, nullptr, pts);

  // 5. cls branch: deform conv on cls_feat, relu -> bufB ; 1x1 -> d_out[0:80]
  deform_gemm<<<gemm_grid, 256, 0, stream>>>(bufA, wpack + (size_t)7 * WSLOT, pts, bufB);
  conv1x1<<<dim3(NSP / 256, 80), 256, 0, stream>>>(bufB, cls_out_w, cls_out_b,
                                                   out, 0, 80, nullptr, nullptr);

  // 6. refine branch: deform conv on reg_feat -> bufD ; 1x1 + pts -> d_out[98:116]
  deform_gemm<<<gemm_grid, 256, 0, stream>>>(bufC, wpack + (size_t)8 * WSLOT, pts, bufD);
  conv1x1<<<dim3(NSP / 256, 18), 256, 0, stream>>>(bufD, ref_out_w, ref_out_b,
                                                   out, 98, 18, pts, nullptr);
}
